// EmbedVector_30666066494098
// MI455X (gfx1250) — matrix operand order repaired and checked by reading, not compiled, not run on hardware
//
#include <hip/hip_runtime.h>
#include <math.h>

typedef _Float16 half_t;
typedef __attribute__((ext_vector_type(16))) _Float16 v16h;
typedef __attribute__((ext_vector_type(8)))  _Float16 v8h;
typedef __attribute__((ext_vector_type(8)))  float    v8f;

// ---------------------------------------------------------------------------
// WMMA fragment loaders (16x16x32 f16; contiguous k order, same in A and B)
//   A (16xK tile, row-major, lda in halves): lane holds row (lane&15),
//     halves[0:8]  = K + (lane>>4)*16 .. +8
//     halves[8:16] = K + (lane>>4)*16 + 8 .. +16
//   B (Kx16 tile from row-major W (N,K)): lane holds col (lane&15),
//     halves[0:16] = K + (lane>>4)*16 .. +16   (contiguous in W's row)
// ---------------------------------------------------------------------------
__device__ __forceinline__ v16h load_a_frag(const half_t* A, int lda, int k) {
  const int lane = threadIdx.x & 31;
  const int row  = lane & 15;
  const int kb   = (lane >> 4) << 4;
  const half_t* p = A + (size_t)row * lda + (k + kb);
  v8h lo = *(const v8h*)(p);
  v8h hi = *(const v8h*)(p + 8);
  v16h r;
#pragma unroll
  for (int i = 0; i < 8; ++i) { r[i] = lo[i]; r[i + 8] = hi[i]; }
  return r;
}

__device__ __forceinline__ v16h load_b_frag(const half_t* W, int ldb, int k) {
  const int lane = threadIdx.x & 31;
  const int n    = lane & 15;
  const int kb   = (lane >> 4) << 4;
  return *(const v16h*)(W + (size_t)n * ldb + (k + kb));
}

// ---------------------------------------------------------------------------
// C = A(MxK,f16) @ B(NxK,f16)^T + bias1 + bias2 ; one 16x16 tile per wave.
// ---------------------------------------------------------------------------
__global__ void gemm_wmma_bias(const half_t* __restrict__ A, int lda,
                               const half_t* __restrict__ Bm, int ldb,
                               void* __restrict__ Cout, int ldc, int out_f16,
                               const float* __restrict__ bias1,
                               const float* __restrict__ bias2,
                               int M, int N, int K) {
  const int wid = blockIdx.x * (blockDim.x >> 5) + (threadIdx.x >> 5);
  const int tiles_n = N >> 4;
  const int tiles_m = M >> 4;
  if (wid >= tiles_m * tiles_n) return;          // wave-uniform; EXEC stays full
  const int tn = wid % tiles_n;
  const int tm = wid / tiles_n;

  const half_t* Abase = A  + (size_t)(tm * 16) * lda;
  const half_t* Bbase = Bm + (size_t)(tn * 16) * ldb;

  v8f acc = {};
  for (int k = 0; k < K; k += 32) {
    v16h a = load_a_frag(Abase, lda, k);
    v16h b = load_b_frag(Bbase, ldb, k);
    // speculative prefetch of the next A chunk (gfx1250 global_prefetch_b8)
    __builtin_prefetch((const void*)(Abase + (size_t)(threadIdx.x & 15) * lda + k + 64), 0, 0);
    acc = __builtin_amdgcn_wmma_f32_16x16x32_f16(false, a, false, b,
                                                 (short)0, acc, false, false);
  }

  const int lane = threadIdx.x & 31;
  const int mhi  = (lane >> 4) << 3;
  const int n    = tn * 16 + (lane & 15);
  float bs = 0.f;
  if (bias1) bs += bias1[n];
  if (bias2) bs += bias2[n];
#pragma unroll
  for (int r = 0; r < 8; ++r) {
    const int m = tm * 16 + mhi + r;
    const float v = acc[r] + bs;
    if (out_f16) ((half_t*)Cout)[(size_t)m * ldc + n] = (half_t)v;
    else         ((float*)Cout)[(size_t)m * ldc + n]  = v;
  }
}

// ---------------------------------------------------------------------------
// Fused LSTM step: g = xg[t] + h_prev @ Whh^T ; cell update ; h -> feat (f16).
// One wave = 16 batch rows x 16 hidden cols, 4 gate accumulators.
// Grid: 32x16 = 512 tiles = 64 blocks x 8 waves.
// ---------------------------------------------------------------------------
__global__ void lstm_step(const half_t* __restrict__ xg,   // (T*B, 1024) f16
                          const half_t* __restrict__ whh,  // (1024, 256) f16
                          half_t* __restrict__ feat,       // (T*B, 832) f16
                          float*  __restrict__ cst,        // (512, 256) f32
                          int t, int tprev, int colOff, int has_hprev) {
  const int wid  = blockIdx.x * (blockDim.x >> 5) + (threadIdx.x >> 5);
  const int tn   = wid & 15;   // hidden tile
  const int tm   = wid >> 4;   // batch tile
  const int lane = threadIdx.x & 31;

  v8f aI = {}, aF = {}, aG = {}, aO = {};
  if (has_hprev) {
    const half_t* Abase = feat + (size_t)(tprev * 512 + tm * 16) * 832 + colOff;
    const half_t* BI = whh + (size_t)(0   + tn * 16) * 256;
    const half_t* BF = whh + (size_t)(256 + tn * 16) * 256;
    const half_t* BG = whh + (size_t)(512 + tn * 16) * 256;
    const half_t* BO = whh + (size_t)(768 + tn * 16) * 256;
    for (int k = 0; k < 256; k += 32) {
      v16h a  = load_a_frag(Abase, 832, k);
      v16h bi = load_b_frag(BI, 256, k);
      v16h bf = load_b_frag(BF, 256, k);
      v16h bg = load_b_frag(BG, 256, k);
      v16h bo = load_b_frag(BO, 256, k);
      aI = __builtin_amdgcn_wmma_f32_16x16x32_f16(false, a, false, bi, (short)0, aI, false, false);
      aF = __builtin_amdgcn_wmma_f32_16x16x32_f16(false, a, false, bf, (short)0, aF, false, false);
      aG = __builtin_amdgcn_wmma_f32_16x16x32_f16(false, a, false, bg, (short)0, aG, false, false);
      aO = __builtin_amdgcn_wmma_f32_16x16x32_f16(false, a, false, bo, (short)0, aO, false, false);
    }
  }

  const int mhi = (lane >> 4) << 3;
  const int n   = tn * 16 + (lane & 15);
#pragma unroll
  for (int r = 0; r < 8; ++r) {
    const int m = tm * 16 + mhi + r;            // batch index
    const size_t row = (size_t)t * 512 + m;     // time-flattened row
    const size_t xgi = row * 1024;
    const float gi = aI[r] + (float)xg[xgi + n];
    const float gf = aF[r] + (float)xg[xgi + 256 + n];
    const float gg = aG[r] + (float)xg[xgi + 512 + n];
    const float go = aO[r] + (float)xg[xgi + 768 + n];
    const float c0 = has_hprev ? cst[(size_t)m * 256 + n] : 0.f;
    const float si = 1.f / (1.f + expf(-gi));
    const float sf = 1.f / (1.f + expf(-gf));
    const float so = 1.f / (1.f + expf(-go));
    const float cn = sf * c0 + si * tanhf(gg);
    cst[(size_t)m * 256 + n] = cn;
    const float h = so * tanhf(cn);
    feat[row * 832 + colOff + n] = (half_t)h;
  }
}

// ---------------------------------------------------------------------------
// Prep / epilogue kernels
// ---------------------------------------------------------------------------
__global__ void embed_feat(const int* __restrict__ text,
                           const float* __restrict__ ew,
                           half_t* __restrict__ feat) {
  const int idx = blockIdx.x * blockDim.x + threadIdx.x;
  if (idx >= 32768 * 320) return;
  const int row = idx / 320, col = idx - row * 320;
  float v = 0.f;
  if (col < 300) v = ew[(size_t)text[row] * 300 + col];
  feat[(size_t)row * 832 + col] = (half_t)v;
}

__global__ void pack_pad(const float* __restrict__ src, half_t* __restrict__ dst,
                         int rows, int ksrc, int kdst) {
  const int idx = blockIdx.x * blockDim.x + threadIdx.x;
  if (idx >= rows * kdst) return;
  const int r = idx / kdst, c = idx - r * kdst;
  dst[idx] = (half_t)((c < ksrc) ? src[(size_t)r * ksrc + c] : 0.f);
}

__global__ void pack_fc1(const float* __restrict__ src, half_t* __restrict__ dst) {
  const int idx = blockIdx.x * blockDim.x + threadIdx.x;
  if (idx >= 512 * 832) return;
  const int r = idx / 832, c = idx - r * 832;
  float v = 0.f;
  if (c < 300)       v = src[(size_t)r * 812 + c];
  else if (c >= 320) v = src[(size_t)r * 812 + (c - 20)];
  dst[idx] = (half_t)v;
}

__global__ void bn_stats(const float* __restrict__ h1,
                         float* __restrict__ mu, float* __restrict__ rstd) {
  const int j = blockIdx.x * blockDim.x + threadIdx.x;
  if (j >= 512) return;
  float s = 0.f, s2 = 0.f;
  for (int r = 0; r < 32768; ++r) {
    const float v = h1[(size_t)r * 512 + j];
    s += v; s2 += v * v;
  }
  const float m   = s * (1.f / 32768.f);
  const float var = s2 * (1.f / 32768.f) - m * m;
  mu[j]   = m;
  rstd[j] = rsqrtf(var + 1e-5f);
}

__global__ void bn_norm_tanh(float* __restrict__ h1,
                             const float* __restrict__ mu,
                             const float* __restrict__ rstd,
                             const float* __restrict__ g,
                             const float* __restrict__ b) {
  const size_t idx = (size_t)blockIdx.x * blockDim.x + threadIdx.x;
  if (idx >= (size_t)32768 * 512) return;
  const int j = (int)(idx & 511);
  h1[idx] = tanhf((h1[idx] - mu[j]) * rstd[j] * g[j] + b[j]);
}

__global__ void time_mean(const float* __restrict__ y, float* __restrict__ hbar) {
  const int idx = blockIdx.x * blockDim.x + threadIdx.x;
  if (idx >= 512 * 512) return;
  const int b = idx >> 9, j = idx & 511;
  float s = 0.f;
  for (int t = 0; t < 64; ++t) s += y[((size_t)t * 512 + b) * 512 + j];
  hbar[idx] = s * (1.f / 64.f);
}

__global__ void fc2_norm(const float* __restrict__ hbar,
                         const float* __restrict__ w2,
                         const float* __restrict__ b2,
                         float* __restrict__ out) {
  __shared__ float row[512];
  __shared__ float sc[64];
  __shared__ float nrm;
  const int b = blockIdx.x, tid = threadIdx.x;   // 64 threads
  for (int i = tid; i < 512; i += 64) row[i] = hbar[(size_t)b * 512 + i];
  __syncthreads();
  if (tid < 50) {
    float s = b2[tid];
    for (int k = 0; k < 512; ++k) s += row[k] * w2[(size_t)tid * 512 + k];
    sc[tid] = s;
  }
  __syncthreads();
  if (tid == 0) {
    float s = 0.f;
    for (int l = 0; l < 50; ++l) s += sc[l] * sc[l];
    nrm = fmaxf(sqrtf(s), 1e-12f);
  }
  __syncthreads();
  if (tid < 50) out[(size_t)b * 50 + tid] = sc[tid] / nrm;
}

// ---------------------------------------------------------------------------
extern "C" void kernel_launch(void* const* d_in, const int* in_sizes, int n_in,
                              void* d_out, int out_size, void* d_ws, size_t ws_size,
                              hipStream_t stream) {
  const int*   text    = (const int*)  d_in[0];
  const float* embed_w = (const float*)d_in[1];
  const float* w_ih_f  = (const float*)d_in[2];
  const float* w_hh_f  = (const float*)d_in[3];
  const float* b_ih_f  = (const float*)d_in[4];
  const float* b_hh_f  = (const float*)d_in[5];
  const float* w_ih_b  = (const float*)d_in[6];
  const float* w_hh_b  = (const float*)d_in[7];
  const float* b_ih_b  = (const float*)d_in[8];
  const float* b_hh_b  = (const float*)d_in[9];
  // d_in[10..13] = attention params: provably dead (softmax over axis=1 of (N,1) == 1)
  const float* fc1_w   = (const float*)d_in[14];
  const float* fc1_b   = (const float*)d_in[15];
  const float* bn_g    = (const float*)d_in[16];
  const float* bn_b    = (const float*)d_in[17];
  const float* fc2_w   = (const float*)d_in[18];
  const float* fc2_b   = (const float*)d_in[19];
  float* out = (float*)d_out;

  const int T = 64, B = 512, D = 300, H = 256;
  const int N = T * B;                 // 32768
  const int KX = 320, G4 = 1024, H2 = 512, KC = 832;

  char* ws = (char*)d_ws;
  size_t cur = 0;
  auto alloc = [&](size_t bytes) -> char* {
    char* p = ws + cur;
    cur = (cur + bytes + 255) & ~((size_t)255);
    return p;
  };
  half_t* feat    = (half_t*)alloc((size_t)N * 832 * 2);
  half_t* xgf     = (half_t*)alloc((size_t)N * 1024 * 2);
  half_t* xgb     = (half_t*)alloc((size_t)N * 1024 * 2);
  half_t* wihf_pk = (half_t*)alloc((size_t)G4 * KX * 2);
  half_t* wihb_pk = (half_t*)alloc((size_t)G4 * KX * 2);
  half_t* whhf_pk = (half_t*)alloc((size_t)G4 * H * 2);
  half_t* whhb_pk = (half_t*)alloc((size_t)G4 * H * 2);
  half_t* fc1_pk  = (half_t*)alloc((size_t)H2 * KC * 2);
  float*  cf      = (float*)alloc((size_t)B * H * 4);
  float*  cb      = (float*)alloc((size_t)B * H * 4);
  float*  h1      = (float*)alloc((size_t)N * H2 * 4);
  float*  mu      = (float*)alloc(512 * 4);
  float*  rstd    = (float*)alloc(512 * 4);
  float*  hbar    = (float*)alloc((size_t)B * H2 * 4);

  // --- prep: embedding gather -> f16 feature buffer; pack weights to padded f16
  embed_feat<<<(N * KX) / 256, 256, 0, stream>>>(text, embed_w, feat);
  pack_pad<<<(G4 * KX + 255) / 256, 256, 0, stream>>>(w_ih_f, wihf_pk, G4, D, KX);
  pack_pad<<<(G4 * KX + 255) / 256, 256, 0, stream>>>(w_ih_b, wihb_pk, G4, D, KX);
  pack_pad<<<(G4 * H + 255) / 256, 256, 0, stream>>>(w_hh_f, whhf_pk, G4, H, H);
  pack_pad<<<(G4 * H + 255) / 256, 256, 0, stream>>>(w_hh_b, whhb_pk, G4, H, H);
  pack_fc1<<<(H2 * KC + 255) / 256, 256, 0, stream>>>(fc1_w, fc1_pk);

  // --- input projections: xg = x @ Wih^T + (b_ih + b_hh), WMMA f16->f32
  {
    const int tiles = (N >> 4) * (G4 >> 4);
    gemm_wmma_bias<<<(tiles + 7) / 8, 256, 0, stream>>>(
        feat, 832, wihf_pk, KX, (void*)xgf, G4, 1, b_ih_f, b_hh_f, N, G4, KX);
    gemm_wmma_bias<<<(tiles + 7) / 8, 256, 0, stream>>>(
        feat, 832, wihb_pk, KX, (void*)xgb, G4, 1, b_ih_b, b_hh_b, N, G4, KX);
  }

  // --- recurrence: 64 fused WMMA+pointwise steps per direction
  for (int s = 0; s < T; ++s) {
    lstm_step<<<64, 256, 0, stream>>>(xgf, whhf_pk, feat, cf, s, s - 1, 320, s > 0);
    const int t = T - 1 - s;
    lstm_step<<<64, 256, 0, stream>>>(xgb, whhb_pk, feat, cb, t, t + 1, 576, s > 0);
  }

  // --- fc1 over the full fused feature (attention is identity): f32 out
  {
    const int tiles = (N >> 4) * (H2 >> 4);
    gemm_wmma_bias<<<(tiles + 7) / 8, 256, 0, stream>>>(
        feat, 832, fc1_pk, KC, (void*)h1, H2, 0, fc1_b, nullptr, N, H2, KC);
  }

  // --- BatchNorm (training-mode batch stats, biased var) + tanh, in place
  bn_stats<<<2, 256, 0, stream>>>(h1, mu, rstd);
  bn_norm_tanh<<<(int)(((size_t)N * H2) / 256), 256, 0, stream>>>(h1, mu, rstd, bn_g, bn_b);

  // --- mean over T commuted through fc2; then fc2 + row L2-normalize
  time_mean<<<(512 * 512) / 256, 256, 0, stream>>>(h1, hbar);
  fc2_norm<<<512, 64, 0, stream>>>(hbar, fc2_w, fc2_b, out);
}
